// HSEM_34248069218878
// MI455X (gfx1250) — matrix operand order repaired and checked by reading, not compiled, not run on hardware
//
#include <hip/hip_runtime.h>
#include <hip/hip_bf16.h>

// ---------------------------------------------------------------------------
// Problem constants (from reference): b=2, c=256, h=w=64 -> n=4096, c_out=512
// ---------------------------------------------------------------------------
#define NB   2
#define C    256
#define N    4096          // 64*64
#define COUT 512
#define P2   1024          // 32*32 downsampled spatial

typedef __attribute__((ext_vector_type(16))) __bf16 v16bf;
typedef __attribute__((ext_vector_type(8)))  float  v8f;

union V16 {
    v16bf v;
    uint4 q[2];
    unsigned short u[16];
};

__device__ __forceinline__ unsigned short f32_to_bf16(float f) {
    unsigned u = __float_as_uint(f);
    unsigned r = 0x7FFFu + ((u >> 16) & 1u);   // round-to-nearest-even
    return (unsigned short)((u + r) >> 16);
}

__device__ __forceinline__ v8f wmma_bf16(const V16& a, const V16& b, v8f c) {
    return __builtin_amdgcn_wmma_f32_16x16x32_bf16(
        false, a.v, false, b.v, (short)0, c, false, false);
}

// ---------------------------------------------------------------------------
// Workspace layout (bytes)
// ---------------------------------------------------------------------------
static const size_t OFF_XF   = 0;                               // ushort NB*N*C (4 MB)   xf bf16, [b][p][ch]
static const size_t OFF_FCW  = OFF_XF   + (size_t)NB*N*C*2;     // ushort C*C    (128 KB) fc_w bf16 [d][c]
static const size_t OFF_OUTW = OFF_FCW  + (size_t)C*C*2;        // ushort COUT*C (256 KB) out_w bf16 [co][ci]
static const size_t OFF_SQ   = OFF_OUTW + (size_t)COUT*C*2;     // float  NB*N   (32 KB)  row squared norms
static const size_t OFF_BITS = OFF_SQ   + (size_t)NB*N*4;       // u32    NB*N*128 (4 MB) H bitmask, 128 words/row
static const size_t OFF_CNT  = OFF_BITS + (size_t)NB*N*128*4;   // int    NB*N   (32 KB)  row degree
static const size_t OFF_INV  = OFF_CNT  + (size_t)NB*N*4;       // float  NB*N   (32 KB)  1/degree
static const size_t OFF_YT   = OFF_INV  + (size_t)NB*N*4;       // ushort NB*C*N (4 MB)   y bf16 transposed [b][d][v]
static const size_t OFF_ET   = OFF_YT   + (size_t)NB*C*N*2;     // ushort NB*C*N (4 MB)   E bf16 transposed [b][d][e]
static const size_t OFF_XR   = OFF_ET   + (size_t)NB*C*N*2;     // float  NB*C*N (8 MB)   residual, NCHW
static const size_t OFF_BNS  = OFF_XR   + (size_t)NB*C*N*4;     // float  2*C    (2 KB)   bn scale/shift
static const size_t OFF_XA   = OFF_BNS  + (size_t)2*C*4;        // float  NB*C*N (8 MB)   post-SiLU, NCHW
static const size_t OFF_XDT  = OFF_XA   + (size_t)NB*C*N*4;     // ushort NB*P2*C (1 MB)  down-conv out bf16 [b][p][ci]

// ---------------------------------------------------------------------------
// K0a: x (NCHW f32) -> xf bf16 [b][p][ch]
// ---------------------------------------------------------------------------
__global__ void k_cvt_xf(const float* __restrict__ x, unsigned short* __restrict__ xf) {
    size_t i = (size_t)blockIdx.x * blockDim.x + threadIdx.x;   // NB*N*C
    int ch = (int)(i & (C - 1));
    size_t rem = i >> 8;
    int p  = (int)(rem & (N - 1));
    int bb = (int)(rem >> 12);
    xf[i] = f32_to_bf16(x[((size_t)bb * C + ch) * N + p]);
}

// K0b: fc_w, out_w -> bf16
__global__ void k_cvt_w(const float* __restrict__ fcwf, const float* __restrict__ outwf,
                        unsigned short* __restrict__ fcw, unsigned short* __restrict__ outw) {
    size_t i = (size_t)blockIdx.x * blockDim.x + threadIdx.x;   // C*C + COUT*C
    if (i < (size_t)C * C) fcw[i] = f32_to_bf16(fcwf[i]);
    else                   outw[i - (size_t)C * C] = f32_to_bf16(outwf[i - (size_t)C * C]);
}

// K0c: per-row squared norms + zero degree counters
__global__ void k_sq_cnt(const float* __restrict__ x, float* __restrict__ sq, int* __restrict__ cnt) {
    int t = blockIdx.x * blockDim.x + threadIdx.x;              // NB*N
    int bb = t >> 12, p = t & (N - 1);
    float s = 0.f;
    const float* src = x + (size_t)bb * C * N + p;
    for (int ch = 0; ch < C; ++ch) { float v = src[(size_t)ch * N]; s += v * v; }
    sq[t] = s;
    cnt[t] = 0;
}

// ---------------------------------------------------------------------------
// K1: Gram via WMMA bf16, 2x2 register blocking (32x32 output per wave),
// threshold -> bitmask H (one u32 word per row) + row degrees.
// NB*128*128 = 32768 waves.
// ---------------------------------------------------------------------------
__global__ void k_gram(const unsigned short* __restrict__ xf, const float* __restrict__ sq,
                       unsigned* __restrict__ bitsw, int* __restrict__ cnt) {
    int wave = threadIdx.x >> 5, lane = threadIdx.x & 31;
    int tile = blockIdx.x * 4 + wave;
    int bb  = tile >> 14;
    int rem = tile & 16383;
    int It = rem >> 7, Jt = rem & 127;       // 32-wide tiles
    int l15 = lane & 15, hi = lane >> 4;

    const unsigned short* xb  = xf + (size_t)bb * N * C;
    const unsigned short* ap0 = xb + ((size_t)(It * 32 + l15)) * C + (hi ? 8 : 0);
    const unsigned short* ap1 = ap0 + (size_t)16 * C;
    const unsigned short* bp0 = xb + ((size_t)(Jt * 32 + l15)) * C + (hi ? 8 : 0);
    const unsigned short* bp1 = bp0 + (size_t)16 * C;

    v8f acc00 = {}, acc01 = {}, acc10 = {}, acc11 = {};
#pragma unroll
    for (int kk = 0; kk < 8; ++kk) {
        V16 a0, a1, b0, b1;
        a0.q[0] = *(const uint4*)(ap0 + kk * 32);
        a0.q[1] = *(const uint4*)(ap0 + kk * 32 + 16);
        a1.q[0] = *(const uint4*)(ap1 + kk * 32);
        a1.q[1] = *(const uint4*)(ap1 + kk * 32 + 16);
        b0.q[0] = *(const uint4*)(bp0 + kk * 32);
        b0.q[1] = *(const uint4*)(bp0 + kk * 32 + 16);
        b1.q[0] = *(const uint4*)(bp1 + kk * 32);
        b1.q[1] = *(const uint4*)(bp1 + kk * 32 + 16);
        acc00 = wmma_bf16(a0, b0, acc00);
        acc01 = wmma_bf16(a0, b1, acc01);
        acc10 = wmma_bf16(a1, b0, acc10);
        acc11 = wmma_bf16(a1, b1, acc11);
    }

    const float* sqb = sq + bb * N;
    unsigned* bw = bitsw + (size_t)bb * N * 128;
    int* cntb = cnt + bb * N;
#pragma unroll
    for (int a = 0; a < 2; ++a) {
        v8f& c0 = a ? acc10 : acc00;
        v8f& c1 = a ? acc11 : acc01;
#pragma unroll
        for (int r = 0; r < 8; ++r) {
            int m  = It * 32 + a * 16 + r + (hi ? 8 : 0);
            int n0 = Jt * 32 + l15;
            float sm = sqb[m];
            float d20 = sm + sqb[n0]      - 2.0f * c0[r];
            float d21 = sm + sqb[n0 + 16] - 2.0f * c1[r];
            unsigned m0 = __builtin_amdgcn_ballot_w32(d20 < 100.0f);  // dist < 10
            unsigned m1 = __builtin_amdgcn_ballot_w32(d21 < 100.0f);
            if (lane == 0) {
                int mlo = It * 32 + a * 16 + r, mhi = mlo + 8;
                unsigned wlo = (m0 & 0xFFFFu) | (m1 << 16);
                unsigned whi = (m0 >> 16)     | (m1 & 0xFFFF0000u);
                bw[(size_t)mlo * 128 + Jt] = wlo;
                bw[(size_t)mhi * 128 + Jt] = whi;
                atomicAdd(&cntb[mlo], __popc(wlo));
                atomicAdd(&cntb[mhi], __popc(whi));
            }
        }
    }
}

// K1b: degree -> reciprocal
__global__ void k_rowinv(const int* __restrict__ cnt, float* __restrict__ inv) {
    int t = blockIdx.x * blockDim.x + threadIdx.x;
    int c = cnt[t];
    inv[t] = (c > 0) ? (1.0f / (float)c) : 0.0f;
}

// ---------------------------------------------------------------------------
// K2: y = xf @ fc_w^T + fc_b  (WMMA), stored bf16 transposed [b][d][v]
// ---------------------------------------------------------------------------
__global__ void k_fc(const unsigned short* __restrict__ xf, const unsigned short* __restrict__ fcw,
                     const float* __restrict__ fcb, unsigned short* __restrict__ yt) {
    int wave = threadIdx.x >> 5, lane = threadIdx.x & 31;
    int tile = blockIdx.x * 4 + wave;          // NB*256*16 = 8192
    int bb  = tile >> 12;
    int rem = tile & 4095;
    int Mt = rem >> 4, Nt = rem & 15;
    int l15 = lane & 15, hi = lane >> 4;

    const unsigned short* ap = xf + ((size_t)bb * N + Mt * 16 + l15) * C + (hi ? 8 : 0);
    const unsigned short* bp = fcw + (size_t)(Nt * 16 + l15) * C + (hi ? 8 : 0);

    v8f acc = {};
#pragma unroll
    for (int kk = 0; kk < 8; ++kk) {
        V16 a, b;
        a.q[0] = *(const uint4*)(ap + kk * 32);
        a.q[1] = *(const uint4*)(ap + kk * 32 + 16);
        b.q[0] = *(const uint4*)(bp + kk * 32);
        b.q[1] = *(const uint4*)(bp + kk * 32 + 16);
        acc = wmma_bf16(a, b, acc);
    }
#pragma unroll
    for (int r = 0; r < 8; ++r) {
        int m = Mt * 16 + r + (hi ? 8 : 0);
        int d = Nt * 16 + l15;
        yt[((size_t)bb * C + d) * N + m] = f32_to_bf16(acc[r] + fcb[d]);
    }
}

// ---------------------------------------------------------------------------
// Masked aggregation, 1x4 N-blocking: one bit-expanded A tile feeds 4 WMMAs.
// Output: 16 rows x 64 cols per wave.
// ---------------------------------------------------------------------------
struct Acc4 { v8f c[4]; };

__device__ __forceinline__ Acc4 agg_tile4(const unsigned* __restrict__ bits,
                                          const unsigned short* __restrict__ Bt,
                                          int bb, int Mt, int Ng, int lane) {
    int l15 = lane & 15, hi = lane >> 4;
    const unsigned* wrd = bits + ((size_t)bb * N + Mt * 16 + l15) * 128;
    const unsigned short* bp0 = Bt + ((size_t)bb * C + Ng * 64 + l15) * N + (hi ? 16 : 0);
    const unsigned short* bp1 = bp0 + (size_t)16 * N;
    const unsigned short* bp2 = bp0 + (size_t)32 * N;
    const unsigned short* bp3 = bp0 + (size_t)48 * N;
    int kbase = hi ? 16 : 0;

    Acc4 A = {};
    for (int kk = 0; kk < 128; ++kk) {
        unsigned w = wrd[kk];
        V16 a;
#pragma unroll
        for (int e = 0; e < 16; ++e) {
            int kloc = e + kbase;
            a.u[e] = (unsigned short)(((w >> kloc) & 1u) * 0x3F80u);   // bf16 1.0 / 0.0
        }
        __builtin_prefetch((const void*)(bp0 + (size_t)(kk + 8) * 32), 0, 1);
        V16 b;
        b.q[0] = *(const uint4*)(bp0 + (size_t)kk * 32);
        b.q[1] = *(const uint4*)(bp0 + (size_t)kk * 32 + 8);
        A.c[0] = wmma_bf16(a, b, A.c[0]);
        b.q[0] = *(const uint4*)(bp1 + (size_t)kk * 32);
        b.q[1] = *(const uint4*)(bp1 + (size_t)kk * 32 + 8);
        A.c[1] = wmma_bf16(a, b, A.c[1]);
        b.q[0] = *(const uint4*)(bp2 + (size_t)kk * 32);
        b.q[1] = *(const uint4*)(bp2 + (size_t)kk * 32 + 8);
        A.c[2] = wmma_bf16(a, b, A.c[2]);
        b.q[0] = *(const uint4*)(bp3 + (size_t)kk * 32);
        b.q[1] = *(const uint4*)(bp3 + (size_t)kk * 32 + 8);
        A.c[3] = wmma_bf16(a, b, A.c[3]);
    }
    return A;
}

// K3: E = inv * (H @ y), stored bf16 transposed [b][d][e]
__global__ void k_e2v(const unsigned* __restrict__ bits, const float* __restrict__ inv,
                      const unsigned short* __restrict__ yt, unsigned short* __restrict__ Et) {
    int wave = threadIdx.x >> 5, lane = threadIdx.x & 31;
    int tile = blockIdx.x * 4 + wave;          // NB*256*4 = 2048
    int bb  = tile >> 10;
    int rem = tile & 1023;
    int Mt = rem >> 2, Ng = rem & 3;
    Acc4 A = agg_tile4(bits, yt, bb, Mt, Ng, lane);
    int l15 = lane & 15, hi = lane >> 4;
#pragma unroll
    for (int j = 0; j < 4; ++j) {
#pragma unroll
        for (int r = 0; r < 8; ++r) {
            int m = Mt * 16 + r + (hi ? 8 : 0);
            int d = Ng * 64 + j * 16 + l15;
            float val = inv[bb * N + m] * A.c[j][r];
            Et[((size_t)bb * C + d) * N + m] = f32_to_bf16(val);
        }
    }
}

// K4: v = inv * (H @ E) + x, stored f32 NCHW
__global__ void k_v2v(const unsigned* __restrict__ bits, const float* __restrict__ inv,
                      const unsigned short* __restrict__ Et, const float* __restrict__ x,
                      float* __restrict__ xr) {
    int wave = threadIdx.x >> 5, lane = threadIdx.x & 31;
    int tile = blockIdx.x * 4 + wave;          // NB*256*4 = 2048
    int bb  = tile >> 10;
    int rem = tile & 1023;
    int Mt = rem >> 2, Ng = rem & 3;
    Acc4 A = agg_tile4(bits, Et, bb, Mt, Ng, lane);
    int l15 = lane & 15, hi = lane >> 4;
#pragma unroll
    for (int j = 0; j < 4; ++j) {
#pragma unroll
        for (int r = 0; r < 8; ++r) {
            int m = Mt * 16 + r + (hi ? 8 : 0);
            int d = Ng * 64 + j * 16 + l15;
            size_t idx = ((size_t)bb * C + d) * N + m;
            xr[idx] = inv[bb * N + m] * A.c[j][r] + x[idx];
        }
    }
}

// ---------------------------------------------------------------------------
// K5: BatchNorm batch statistics (biased var) -> fused scale/shift
// ---------------------------------------------------------------------------
__global__ void k_bnstats(const float* __restrict__ xr, const float* __restrict__ gamma,
                          const float* __restrict__ beta, float* __restrict__ bns) {
    __shared__ float sh[256], sh2[256];
    int ch = blockIdx.x, tid = threadIdx.x;
    float s = 0.f, s2 = 0.f;
    for (int bb = 0; bb < NB; ++bb) {
        const float* p = xr + ((size_t)bb * C + ch) * N;
        for (int i = tid; i < N; i += 256) { float v = p[i]; s += v; s2 += v * v; }
    }
    sh[tid] = s; sh2[tid] = s2;
    __syncthreads();
    for (int o = 128; o > 0; o >>= 1) {
        if (tid < o) { sh[tid] += sh[tid + o]; sh2[tid] += sh2[tid + o]; }
        __syncthreads();
    }
    if (tid == 0) {
        float mean = sh[0] * (1.0f / (NB * N));
        float var  = sh2[0] * (1.0f / (NB * N)) - mean * mean;
        float sc   = gamma[ch] * rsqrtf(var + 1e-5f);
        bns[ch]     = sc;                      // scale
        bns[C + ch] = beta[ch] - mean * sc;    // shift
    }
}

// K6: normalize + SiLU
__global__ void k_act(const float* __restrict__ xr, const float* __restrict__ bns,
                      float* __restrict__ xa) {
    size_t i = (size_t)blockIdx.x * blockDim.x + threadIdx.x;   // NB*C*N
    int ch = (int)((i >> 12) & (C - 1));
    float z = xr[i] * bns[ch] + bns[C + ch];
    xa[i] = z / (1.0f + __expf(-z));
}

// ---------------------------------------------------------------------------
// K7: 3x3 stride-2 conv with right/bottom pad; output stored bf16 transposed
// [b][p2][ci] ready as the WMMA A operand of the 1x1 conv.
// ---------------------------------------------------------------------------
__global__ void k_down(const float* __restrict__ xa, const float* __restrict__ dw,
                       const float* __restrict__ db, unsigned short* __restrict__ xdt) {
    __shared__ float wl[C * 9];
    int blk = blockIdx.x;                      // NB*C = 512
    int bb = blk >> 8, co = blk & (C - 1);
    int tid = threadIdx.x;
    for (int i = tid; i < C * 9; i += 256) wl[i] = dw[(size_t)co * C * 9 + i];
    __syncthreads();
    float bias = db[co];
    for (int j = 0; j < 4; ++j) {
        int p = tid + j * 256;                 // 0..1023
        int oy = p >> 5, ox = p & 31;
        float acc = bias;
        for (int ci = 0; ci < C; ++ci) {
            const float* src = xa + ((size_t)bb * C + ci) * N;
            const float* w = &wl[ci * 9];
#pragma unroll
            for (int ky = 0; ky < 3; ++ky) {
                int iy = 2 * oy + ky;
                if (iy >= 64) continue;
#pragma unroll
                for (int kx = 0; kx < 3; ++kx) {
                    int ix = 2 * ox + kx;
                    if (ix >= 64) continue;
                    acc += src[iy * 64 + ix] * w[ky * 3 + kx];
                }
            }
        }
        xdt[((size_t)bb * P2 + p) * C + co] = f32_to_bf16(acc);
    }
}

// ---------------------------------------------------------------------------
// K8: 1x1 conv as WMMA GEMM: out[b][co][p] = xd[b][p][:] . out_w[co][:] + b
// ---------------------------------------------------------------------------
__global__ void k_out(const unsigned short* __restrict__ xdt, const unsigned short* __restrict__ ow,
                      const float* __restrict__ ob, float* __restrict__ out) {
    int wave = threadIdx.x >> 5, lane = threadIdx.x & 31;
    int tile = blockIdx.x * 4 + wave;          // NB*64*32 = 4096
    int bb  = tile >> 11;
    int rem = tile & 2047;
    int Mt = rem >> 5, Nt = rem & 31;
    int l15 = lane & 15, hi = lane >> 4;

    const unsigned short* ap = xdt + ((size_t)bb * P2 + Mt * 16 + l15) * C + (hi ? 8 : 0);
    const unsigned short* bp = ow + (size_t)(Nt * 16 + l15) * C + (hi ? 8 : 0);

    v8f acc = {};
#pragma unroll
    for (int kk = 0; kk < 8; ++kk) {
        V16 a, b;
        a.q[0] = *(const uint4*)(ap + kk * 32);
        a.q[1] = *(const uint4*)(ap + kk * 32 + 16);
        b.q[0] = *(const uint4*)(bp + kk * 32);
        b.q[1] = *(const uint4*)(bp + kk * 32 + 16);
        acc = wmma_bf16(a, b, acc);
    }
#pragma unroll
    for (int r = 0; r < 8; ++r) {
        int m  = Mt * 16 + r + (hi ? 8 : 0);
        int nc = Nt * 16 + l15;
        out[((size_t)bb * COUT + nc) * P2 + m] = acc[r] + ob[nc];
    }
}

// ---------------------------------------------------------------------------
extern "C" void kernel_launch(void* const* d_in, const int* in_sizes, int n_in,
                              void* d_out, int out_size, void* d_ws, size_t ws_size,
                              hipStream_t stream) {
    (void)in_sizes; (void)n_in; (void)out_size; (void)ws_size;
    const float* x     = (const float*)d_in[0];
    const float* fc_w  = (const float*)d_in[1];
    const float* fc_b  = (const float*)d_in[2];
    const float* gamma = (const float*)d_in[3];
    const float* beta  = (const float*)d_in[4];
    const float* dw    = (const float*)d_in[5];
    const float* db    = (const float*)d_in[6];
    const float* ow    = (const float*)d_in[7];
    const float* ob    = (const float*)d_in[8];
    float* out = (float*)d_out;

    char* W = (char*)d_ws;
    unsigned short* xf   = (unsigned short*)(W + OFF_XF);
    unsigned short* fcw  = (unsigned short*)(W + OFF_FCW);
    unsigned short* outw = (unsigned short*)(W + OFF_OUTW);
    float*          sq   = (float*)(W + OFF_SQ);
    unsigned*       bits = (unsigned*)(W + OFF_BITS);
    int*            cnt  = (int*)(W + OFF_CNT);
    float*          inv  = (float*)(W + OFF_INV);
    unsigned short* yt   = (unsigned short*)(W + OFF_YT);
    unsigned short* Et   = (unsigned short*)(W + OFF_ET);
    float*          xr   = (float*)(W + OFF_XR);
    float*          bns  = (float*)(W + OFF_BNS);
    float*          xa   = (float*)(W + OFF_XA);
    unsigned short* xdt  = (unsigned short*)(W + OFF_XDT);

    k_cvt_xf <<<(NB * N * C) / 256, 256, 0, stream>>>(x, xf);
    k_cvt_w  <<<(C * C + COUT * C) / 256, 256, 0, stream>>>(fc_w, ow, fcw, outw);
    k_sq_cnt <<<(NB * N) / 256, 256, 0, stream>>>(x, sq, cnt);

    k_gram   <<<(NB * 128 * 128) / 4, 128, 0, stream>>>(xf, sq, bits, cnt);
    k_rowinv <<<(NB * N) / 256, 256, 0, stream>>>(cnt, inv);

    k_fc     <<<(NB * 256 * 16) / 4, 128, 0, stream>>>(xf, fcw, fc_b, yt);
    k_e2v    <<<(NB * 256 * 4) / 4, 128, 0, stream>>>(bits, inv, yt, Et);
    k_v2v    <<<(NB * 256 * 4) / 4, 128, 0, stream>>>(bits, inv, Et, x, xr);

    k_bnstats<<<C, 256, 0, stream>>>(xr, gamma, beta, bns);
    k_act    <<<(NB * C * N) / 256, 256, 0, stream>>>(xr, bns, xa);

    k_down   <<<NB * C, 256, 0, stream>>>(xa, dw, db, xdt);
    k_out    <<<(NB * 64 * 32) / 4, 128, 0, stream>>>(xdt, outw, ob, out);
}
